// AttentionOp_68264210202845
// MI455X (gfx1250) — matrix operand order repaired and checked by reading, not compiled, not run on hardware
//
#include <hip/hip_runtime.h>
#include <hip/hip_bf16.h>
#include <math.h>

// ---------------------------------------------------------------------------
// GQA prefill attention + KV cache update for gfx1250 (MI455X), wave32 WMMA.
// Round 3: 6-arg tensor_load_to_lds builtin (clang-23 / therock-10.0 lane).
// ---------------------------------------------------------------------------

typedef __attribute__((ext_vector_type(16))) __bf16 v16bf;
typedef __attribute__((ext_vector_type(8)))  float  v8f;
typedef __attribute__((ext_vector_type(4)))  unsigned int u32x4;
typedef __attribute__((ext_vector_type(8)))  int i32x8;
typedef __attribute__((ext_vector_type(4)))  int i32x4;

#define NUM_HEADS    32
#define HEAD_DIM     128
#define NUM_KV_HEADS 8
#define GQA          4
#define Q_LEN        1024
#define MAX_SEQ      4096
#define CACHE_BATCH  4

#define OUT_ATTN_ELEMS (Q_LEN * NUM_HEADS * HEAD_DIM)                      // 4,194,304
#define CACHE_ELEMS    (CACHE_BATCH * NUM_KV_HEADS * MAX_SEQ * HEAD_DIM)  // 16,777,216
#define KBF_ELEMS      (NUM_KV_HEADS * MAX_SEQ * HEAD_DIM)                // 4,194,304 bf16

#define QTILE 128   // q rows per block: 8 waves x 16 rows
#define KTILE 32    // keys per iteration: 2 x 16-wide WMMA subtiles

// LDS pitches (bf16 elements). Chosen so TDM pad fields can reproduce them:
// K  row = 128 bf16 = 64 dwords, pad 4 dwords  -> pitch 136 bf16 (272B, 16B-mult)
// Vt row = 32  bf16 = 16 dwords, pad 4 dwords  -> pitch 40 bf16  (80B, 16B-mult)
#define LDSK_PITCH 136
#define LDSV_PITCH 40
#define LDSP_PITCH 40

#if __has_builtin(__builtin_amdgcn_tensor_load_to_lds)
#define HAVE_TDM 1
#define NBUF 2
#else
#define HAVE_TDM 0
#define NBUF 1
#endif

// ---------------------------------------------------------------------------
// Kernel 1: KV cache update (pure bandwidth). Single-pass window select.
// ---------------------------------------------------------------------------
__global__ void cache_update_kernel(const float4* __restrict__ past_k,
                                    const float4* __restrict__ past_v,
                                    const float4* __restrict__ new_k,
                                    const float4* __restrict__ new_v,
                                    const int* __restrict__ bp_p,
                                    const int* __restrict__ sp_p,
                                    float4* __restrict__ out_k,
                                    float4* __restrict__ out_v) {
    const int bp = *bp_p;
    const int sp = *sp_p;
    const long idx = (long)blockIdx.x * blockDim.x + threadIdx.x;  // float4 index
    const int HD4 = HEAD_DIM / 4;
    const int  d4  = (int)(idx % HD4);
    const long r1  = idx / HD4;
    const int  m   = (int)(r1 % MAX_SEQ);
    const long r2  = r1 / MAX_SEQ;
    const int  kvh = (int)(r2 % NUM_KV_HEADS);
    const int  b   = (int)(r2 / NUM_KV_HEADS);
    const bool win = (b == bp) && (m >= sp) && (m < sp + Q_LEN);
    const long nidx = ((long)kvh * Q_LEN + (m - sp)) * HD4 + d4;
    out_k[idx] = win ? new_k[nidx] : past_k[idx];
    out_v[idx] = win ? new_v[nidx] : past_v[idx];
}

// ---------------------------------------------------------------------------
// Kernel 2: one-time f32 -> bf16 conversion of the active cache row (batch bp)
//   kbf [kvh][seq][hd]   row-major bf16
//   vtbf[kvh][hd][seq]   transposed bf16 (so PV B-frags are contiguous)
// One block = 32 keys x 128 dims of one kv head.
// ---------------------------------------------------------------------------
__global__ __launch_bounds__(256)
void convert_kv_kernel(const float* __restrict__ kcache,
                       const float* __restrict__ vcache,
                       const int* __restrict__ bp_p,
                       __bf16* __restrict__ kbf,
                       __bf16* __restrict__ vtbf) {
    __shared__ __bf16 ldsT[HEAD_DIM][KTILE + 2];

    const int kvh  = blockIdx.y;
    const int key0 = blockIdx.x * KTILE;
    const int tid  = threadIdx.x;
    const int bp   = *bp_p;

    const float* kb = kcache + ((long)bp * NUM_KV_HEADS + kvh) * (long)MAX_SEQ * HEAD_DIM;
    const float* vb = vcache + ((long)bp * NUM_KV_HEADS + kvh) * (long)MAX_SEQ * HEAD_DIM;
    __bf16* kout = kbf + ((long)kvh * MAX_SEQ + key0) * HEAD_DIM;

    #pragma unroll
    for (int i = 0; i < 4; ++i) {
        const int e4 = (tid + i * 256) * 4;   // 0..4092 step 4
        const int ky = e4 >> 7;
        const int hd = e4 & 127;
        const float4 k4 = *(const float4*)(kb + (long)(key0 + ky) * HEAD_DIM + hd);
        __bf16* kd = kout + (long)ky * HEAD_DIM + hd;
        kd[0] = (__bf16)k4.x; kd[1] = (__bf16)k4.y;
        kd[2] = (__bf16)k4.z; kd[3] = (__bf16)k4.w;
        const float4 v4 = *(const float4*)(vb + (long)(key0 + ky) * HEAD_DIM + hd);
        ldsT[hd + 0][ky] = (__bf16)v4.x;
        ldsT[hd + 1][ky] = (__bf16)v4.y;
        ldsT[hd + 2][ky] = (__bf16)v4.z;
        ldsT[hd + 3][ky] = (__bf16)v4.w;
    }
    __syncthreads();
    // write out transposed V: 128 rows x 32 bf16; thread = (row, 16-key half)
    const int row  = tid >> 1;
    const int kh   = (tid & 1) * 16;
    __bf16* vd = vtbf + ((long)kvh * HEAD_DIM + row) * MAX_SEQ + key0 + kh;
    #pragma unroll
    for (int e = 0; e < 16; ++e) vd[e] = ldsT[row][kh + e];
}

// ---------------------------------------------------------------------------
// TDM helper: issue a 2D tensor_load_to_lds (bf16 elements) with LDS padding.
// D# built per CDNA5 ISA 8.3-8.5: group0 {count=1, lds_addr, global_addr,
// type=2}; group1 {data_size=2B, pad fields, dims/tile/stride}; groups 2/3 = 0.
// ---------------------------------------------------------------------------
#if HAVE_TDM
__device__ __forceinline__ void tdm_load_2d(unsigned int lds_off,
                                            const void* gptr,
                                            unsigned int tensor_d0,
                                            unsigned int tensor_d1,
                                            unsigned int tile_d0,
                                            unsigned int tile_d1,
                                            unsigned long long stride0,
                                            unsigned int pad_interval,  // enc: 3=16dw, 5=64dw
                                            unsigned int pad_amount) {  // enc: N-1 dwords
    const unsigned long long ga = (unsigned long long)(uintptr_t)gptr;
    u32x4 g0;
    g0[0] = 1u;                                             // count=1 (valid user D#)
    g0[1] = lds_off;                                        // lds_addr (bytes)
    g0[2] = (unsigned int)(ga & 0xffffffffu);               // global_addr[31:0]
    g0[3] = (unsigned int)((ga >> 32) & 0x1ffffffu) | (2u << 30);  // [56:32] | type=2
    i32x8 g1;
    g1[0] = (int)((1u << 16) | (1u << 20) |                 // data_size=2B, pad_enable
                  (pad_interval << 22) | (pad_amount << 25));
    g1[1] = (int)((tensor_d0 & 0xffffu) << 16);             // tensor_dim0[15:0]
    g1[2] = (int)(((tensor_d0 >> 16) & 0xffffu) |
                  ((tensor_d1 & 0xffffu) << 16));           // dim0[31:16] | dim1[15:0]
    g1[3] = (int)(((tensor_d1 >> 16) & 0xffffu) |
                  ((tile_d0 & 0xffffu) << 16));             // dim1[31:16] | tile_dim0
    g1[4] = (int)(tile_d1 & 0xffffu);                       // tile_dim1 (tile_dim2=0)
    g1[5] = (int)(stride0 & 0xffffffffu);                   // tensor_dim0_stride[31:0]
    g1[6] = (int)((stride0 >> 32) & 0xffffu);               // stride0[47:32] (dim1_stride lo16=0)
    g1[7] = 0;
    i32x4 gz;
    gz[0] = 0; gz[1] = 0; gz[2] = 0; gz[3] = 0;
    i32x8 gz8;
    gz8[0] = 0; gz8[1] = 0; gz8[2] = 0; gz8[3] = 0;
    gz8[4] = 0; gz8[5] = 0; gz8[6] = 0; gz8[7] = 0;
    __builtin_amdgcn_tensor_load_to_lds(g0, g1, gz, gz, gz8, 0);
}
#endif

// ---------------------------------------------------------------------------
// Kernel 3: flash-attention prefill over the bf16 workspace K/V.
// One block = (128 q rows, 1 head); one wave = 16 q rows x 128 dims.
// ---------------------------------------------------------------------------
__global__ __launch_bounds__(256)
void gqa_prefill_kernel(const float* __restrict__ q_in,
                        const __bf16* __restrict__ kbf,   // [kvh][seq][hd]
                        const __bf16* __restrict__ vtbf,  // [kvh][hd][seq]
                        const int*   __restrict__ sp_p,
                        const float* __restrict__ scale_p,
                        float*       __restrict__ out) {
    __shared__ __bf16 ldsK[NBUF][KTILE][LDSK_PITCH];       // K tile  [key][hd]
    __shared__ __bf16 ldsVt[NBUF][HEAD_DIM][LDSV_PITCH];   // V tile  [hd][key]
    __shared__ __bf16 ldsP[8][16][LDSP_PITCH];             // per-wave P relayout

    const int h      = blockIdx.y;
    const int kvh    = h / GQA;
    const int qblock = blockIdx.x * QTILE;
    const int tid    = threadIdx.x;
    const int wave   = tid >> 5;
    const int lane   = tid & 31;
    const int half   = lane >> 4;
    const int ln     = lane & 15;

    const int   sp    = *sp_p;
    const float scale = *scale_p;

    const __bf16* kbase  = kbf  + (long)kvh * MAX_SEQ * HEAD_DIM;
    const __bf16* vtbase = vtbf + (long)kvh * HEAD_DIM * MAX_SEQ;

    // ---- Q -> 4 A-fragments (16x32 bf16); contiguous k order:
    // lanes 0-15 hold K {0..15}, lanes 16-31 hold K {16..31}.
    const int    qr0  = qblock + wave * 16;
    const float* qrow = q_in + ((long)h * Q_LEN + (qr0 + ln)) * HEAD_DIM;
    v16bf aq[4];
    #pragma unroll
    for (int c = 0; c < 4; ++c) {
        #pragma unroll
        for (int e = 0; e < 16; ++e) {
            const int kl = 16 * half + e;
            aq[c][e] = (__bf16)qrow[32 * c + kl];
        }
    }

    v8f o[8];
    #pragma unroll
    for (int c = 0; c < 8; ++c)
        #pragma unroll
        for (int r = 0; r < 8; ++r) o[c][r] = 0.0f;
    float mrun[8], lrun[8];
    #pragma unroll
    for (int r = 0; r < 8; ++r) { mrun[r] = -3.0e38f; lrun[r] = 0.0f; }

    const int last_q = sp + qblock + QTILE - 1;
    int ntiles = last_q / KTILE + 1;
    if (ntiles > MAX_SEQ / KTILE) ntiles = MAX_SEQ / KTILE;

#if HAVE_TDM
    // prologue: wave 0 launches TDM copy of tile 0 into buffer 0
    if (wave == 0) {
        tdm_load_2d((unsigned int)(uintptr_t)&ldsK[0][0][0], kbase,
                    HEAD_DIM, MAX_SEQ, HEAD_DIM, KTILE, HEAD_DIM, 5u, 3u);
        tdm_load_2d((unsigned int)(uintptr_t)&ldsVt[0][0][0], vtbase,
                    MAX_SEQ, HEAD_DIM, KTILE, HEAD_DIM, MAX_SEQ, 3u, 3u);
    }
#endif

    for (int jt = 0; jt < ntiles; ++jt) {
        const int key0 = jt * KTILE;
        const int buf  = jt & (NBUF - 1);

        __syncthreads();   // everyone done reading the buffer we are about to fill
#if HAVE_TDM
        if (wave == 0) {
            if (jt + 1 < ntiles) {     // overlap: TDM for jt+1 while we compute jt
                const int nb = (jt + 1) & 1;
                tdm_load_2d((unsigned int)(uintptr_t)&ldsK[nb][0][0],
                            kbase + (long)(key0 + KTILE) * HEAD_DIM,
                            HEAD_DIM, MAX_SEQ, HEAD_DIM, KTILE, HEAD_DIM, 5u, 3u);
                tdm_load_2d((unsigned int)(uintptr_t)&ldsVt[nb][0][0],
                            vtbase + key0 + KTILE,
                            MAX_SEQ, HEAD_DIM, KTILE, HEAD_DIM, MAX_SEQ, 3u, 3u);
                __builtin_amdgcn_s_wait_tensorcnt(2);  // tile jt complete
            } else {
                __builtin_amdgcn_s_wait_tensorcnt(0);
            }
        }
#else
        // fallback staging: bf16 global -> LDS, 16B vectors
        {
            const uint4* ks = (const uint4*)(kbase + (long)key0 * HEAD_DIM);
            #pragma unroll
            for (int i = 0; i < 2; ++i) {
                const int v = tid + i * 256;        // 512 x 16B for K tile
                const int ky = v >> 4;              // /16 vec per row
                const int hx = (v & 15) * 8;
                *(uint4*)&ldsK[0][ky][hx] = ks[ky * (HEAD_DIM / 8) + (v & 15)];
            }
            #pragma unroll
            for (int i = 0; i < 2; ++i) {
                const int v = tid + i * 256;        // 512 x 16B for Vt tile
                const int row = v >> 1;             // hd row
                const int kx  = (v & 1) * 16;
                const uint4* vs = (const uint4*)(vtbase + (long)row * MAX_SEQ + key0 + kx);
                *(uint4*)&ldsVt[0][row][kx] = vs[0];
            }
        }
#endif
        __syncthreads();

        // ---- S = Q * K^T : two 16-key subtiles; preload B-frags, then WMMA.
        v8f sarr[2];
        #pragma unroll
        for (int nt = 0; nt < 2; ++nt) {
            v16bf bk[4];
            #pragma unroll
            for (int c = 0; c < 4; ++c)
                #pragma unroll
                for (int e = 0; e < 16; ++e)
                    bk[c][e] = ldsK[buf][nt * 16 + ln][32 * c + 16 * half + e];
            v8f s;
            #pragma unroll
            for (int r = 0; r < 8; ++r) s[r] = 0.0f;
            #pragma unroll
            for (int c = 0; c < 4; ++c)
                s = __builtin_amdgcn_wmma_f32_16x16x32_bf16(
                        false, aq[c], false, bk[c], (short)0, s, false, false);
            sarr[nt] = s;
        }

        // ---- scale, causal mask, online softmax (row r+8*half, col ln)
        const int kp0 = key0 + ln;
        const int kp1 = key0 + 16 + ln;
        float p0[8], p1[8], sf[8];
        #pragma unroll
        for (int r = 0; r < 8; ++r) {
            const int qpos = sp + qr0 + r + 8 * half;
            float v0 = sarr[0][r] * scale;
            float v1 = sarr[1][r] * scale;
            if (kp0 > qpos) v0 = -3.0e38f;
            if (kp1 > qpos) v1 = -3.0e38f;
            float rm = fmaxf(v0, v1);
            #pragma unroll
            for (int msk = 1; msk < 16; msk <<= 1)
                rm = fmaxf(rm, __shfl_xor(rm, msk, 16));
            const float mnew = fmaxf(mrun[r], rm);
            const float sc = (mrun[r] <= -1.0e38f) ? 0.0f : __expf(mrun[r] - mnew);
            const float e0 = (v0 <= -1.0e38f) ? 0.0f : __expf(v0 - mnew);
            const float e1 = (v1 <= -1.0e38f) ? 0.0f : __expf(v1 - mnew);
            float rs = e0 + e1;
            #pragma unroll
            for (int msk = 1; msk < 16; msk <<= 1)
                rs += __shfl_xor(rs, msk, 16);
            lrun[r] = lrun[r] * sc + rs;
            mrun[r] = mnew;
            sf[r]   = sc;
            p0[r]   = e0;
            p1[r]   = e1;
        }
        #pragma unroll
        for (int c = 0; c < 8; ++c)
            #pragma unroll
            for (int r = 0; r < 8; ++r) o[c][r] *= sf[r];

        // ---- P: C-layout -> A-layout via per-wave LDS scratch
        #pragma unroll
        for (int r = 0; r < 8; ++r) {
            ldsP[wave][r + 8 * half][ln]      = (__bf16)p0[r];
            ldsP[wave][r + 8 * half][16 + ln] = (__bf16)p1[r];
        }
        __builtin_amdgcn_wave_barrier();
        v16bf ap;
        #pragma unroll
        for (int e = 0; e < 16; ++e) {
            const int kl = 16 * half + e;
            ap[e] = ldsP[wave][ln][kl];
        }
        __builtin_amdgcn_wave_barrier();

        // ---- O += P * V : preload all 8 Vt B-frags, then the WMMA chain
        v16bf bv[8];
        #pragma unroll
        for (int c = 0; c < 8; ++c)
            #pragma unroll
            for (int e = 0; e < 16; ++e)
                bv[c][e] = ldsVt[buf][16 * c + ln][16 * half + e];
        #pragma unroll
        for (int c = 0; c < 8; ++c)
            o[c] = __builtin_amdgcn_wmma_f32_16x16x32_bf16(
                       false, ap, false, bv[c], (short)0, o[c], false, false);
    }

    // ---- epilogue: normalize and store attn_output[q][h*128 + hd]
    float rcp[8];
    #pragma unroll
    for (int r = 0; r < 8; ++r) rcp[r] = 1.0f / lrun[r];
    #pragma unroll
    for (int c = 0; c < 8; ++c) {
        #pragma unroll
        for (int r = 0; r < 8; ++r) {
            const int row = qr0 + r + 8 * half;
            const int col = h * HEAD_DIM + 16 * c + ln;
            out[(long)row * (NUM_HEADS * HEAD_DIM) + col] = o[c][r] * rcp[r];
        }
    }
}

// ---------------------------------------------------------------------------
extern "C" void kernel_launch(void* const* d_in, const int* in_sizes, int n_in,
                              void* d_out, int out_size, void* d_ws, size_t ws_size,
                              hipStream_t stream) {
    (void)in_sizes; (void)n_in; (void)out_size; (void)ws_size;

    const float* q      = (const float*)d_in[0];
    const float* k_new  = (const float*)d_in[1];
    const float* v_new  = (const float*)d_in[2];
    // d_in[3]: attn_mask — equivalent causal mask recomputed from seq_position
    const int*   bp     = (const int*)d_in[4];
    const float* past_k = (const float*)d_in[5];
    const float* past_v = (const float*)d_in[6];
    const int*   sp     = (const int*)d_in[7];
    const float* scale  = (const float*)d_in[8];

    float* out   = (float*)d_out;
    float* out_k = out + OUT_ATTN_ELEMS;
    float* out_v = out_k + CACHE_ELEMS;

    __bf16* kbf  = (__bf16*)d_ws;             // [8][4096][128] bf16, 8 MB
    __bf16* vtbf = kbf + KBF_ELEMS;           // [8][128][4096] bf16, 8 MB

    // 1) cache update (produces k_cache / v_cache outputs)
    const int nvec4 = CACHE_ELEMS / 4;
    cache_update_kernel<<<nvec4 / 256, 256, 0, stream>>>(
        (const float4*)past_k, (const float4*)past_v,
        (const float4*)k_new,  (const float4*)v_new,
        bp, sp, (float4*)out_k, (float4*)out_v);

    // 2) one-time bf16 conversion of the active cache row (+ V transpose)
    dim3 cgrid(MAX_SEQ / KTILE, NUM_KV_HEADS);
    convert_kv_kernel<<<cgrid, 256, 0, stream>>>(out_k, out_v, bp, kbf, vtbf);

    // 3) flash attention over bf16 workspace
    dim3 grid(Q_LEN / QTILE, NUM_KV_HEADS == 0 ? 1 : NUM_HEADS);
    gqa_prefill_kernel<<<grid, 256, 0, stream>>>(q, kbf, vtbf, sp, scale, out);
}
